// MyRNN_2920577761470
// MI455X (gfx1250) — compile-verified, matrix operand order checked by reading, not run on hardware
//
#include <hip/hip_runtime.h>
#include <hip/hip_bf16.h>

// ---------------------------------------------------------------------------
// 2-layer LSTM (B=512, T=80, E=256, H1=H2=512) for MI455X (gfx1250).
// bf16 WMMA (v_wmma_f32_16x16x32_bf16) with fp32 accumulation.
// - Gate-fused GEMM: each block computes all 4 gates for a j-slice; the LSTM
//   cell update runs in the GEMM epilogue (no z buffer, no pointwise pass).
// - Double-buffered LDS with register-staged prefetch: one barrier per chunk,
//   next chunk's global loads overlap current chunk's WMMAs.
// - h double-buffered across timesteps; c updated in place.
// ---------------------------------------------------------------------------

typedef __attribute__((ext_vector_type(16))) __bf16 v16bf;
typedef __attribute__((ext_vector_type(8)))  float  v8f;

#define BQ  512      // batch
#define TT  80       // timesteps
#define EE  256      // embedding dim
#define HH  512      // hidden dim (both layers)
#define NB  2048     // 4*H gate width

union FragU { unsigned int u[8]; v16bf v; };

__device__ __forceinline__ unsigned int pack_bf2(float lo, float hi) {
    // round-to-nearest-even f32 -> bf16, packed pair
    unsigned int a = __float_as_uint(lo);
    unsigned int b = __float_as_uint(hi);
    a = (a + 0x7FFFu + ((a >> 16) & 1u)) >> 16;
    b = (b + 0x7FFFu + ((b >> 16) & 1u)) >> 16;
    return (b << 16) | a;
}

__device__ __forceinline__ float sigmoidf_(float x) {
    return 1.f / (1.f + __expf(-x));
}

// pack 16 floats (4x float4) -> 8 bf16-pair dwords, store as 2x b128 to LDS
__device__ __forceinline__ void pack_store_a(const float4 f[4], unsigned int* dst)
{
    unsigned int pk[8];
    pk[0] = pack_bf2(f[0].x, f[0].y); pk[1] = pack_bf2(f[0].z, f[0].w);
    pk[2] = pack_bf2(f[1].x, f[1].y); pk[3] = pack_bf2(f[1].z, f[1].w);
    pk[4] = pack_bf2(f[2].x, f[2].y); pk[5] = pack_bf2(f[2].z, f[2].w);
    pk[6] = pack_bf2(f[3].x, f[3].y); pk[7] = pack_bf2(f[3].z, f[3].w);
    ((uint4*)dst)[0] = make_uint4(pk[0], pk[1], pk[2], pk[3]);
    ((uint4*)dst)[1] = make_uint4(pk[4], pk[5], pk[6], pk[7]);
}

// ---------------------------------------------------------------------------
// Fused LSTM step (one layer):
//   z[b, g*512+j] = A[b, :] . WT[g*512+j, :]   (bf16 WMMA, fp32 accum)
//   A columns [0,K0)   : embeddings (tokens gather) if USE_EMB, else x0
//   A columns [K0,KTOT): hprev
//   epilogue: c = sig(f)*c + sig(i)*tanh(g); hOut = sig(o)*tanh(c)
//
// Block: 256 threads (8 waves). Tile: 128 batch rows x 32 j (x 4 gates).
// Wave: 32 rows x 16 j; its 4 N-fragments are the 4 gates of those 16 j.
// Grid: (HH/32 = 16, BQ/128 = 4) = 64 workgroups.
// ---------------------------------------------------------------------------
template<int KTOT, int K0, bool USE_EMB>
__global__ __launch_bounds__(256)
void lstm_step(const int*   __restrict__ tokens,   // [B, T]   (USE_EMB only)
               const float* __restrict__ emb,      // [V, E]   (USE_EMB only)
               const float* __restrict__ x0,       // [B, K0]  (!USE_EMB)
               const float* __restrict__ hprev,    // [B, KTOT-K0]
               const unsigned short* __restrict__ WT, // [NB, KTOT] bf16 bits
               const float* __restrict__ bias,     // [NB]
               float* __restrict__ cState,         // [B, HH] in-place
               float* __restrict__ hOut,           // [B, HH] (other buffer)
               int t)
{
    // double-buffered staging: [2][128 rows][16 dwords] (32 bf16/row)
    __shared__ __align__(16) unsigned int As[2 * 128 * 16];
    __shared__ __align__(16) unsigned int Bs[2 * 128 * 16];

    const int tid  = threadIdx.x;
    const int lane = tid & 31;
    const int wave = tid >> 5;           // 0..7
    const int wm   = wave & 3;           // 4 waves along M (batch)
    const int wn   = wave >> 2;          // 2 waves along j
    const int j0   = blockIdx.x * 32;    // j-slice base
    const int m0   = blockIdx.y * 128;   // batch-row tile

    const int srow  = tid >> 1;          // staging row 0..127
    const int shalf = tid & 1;           // which 16-element half of the chunk
    constexpr int K1      = KTOT - K0;
    constexpr int C0      = K0 >> 5;     // chunks sourced from phase-0
    constexpr int NCHUNK  = KTOT >> 5;

    // ---- loop-invariant source pointers (token gather hoisted) ----
    const float* aRow0;
    if (USE_EMB) {
        const int tok = tokens[(m0 + srow) * TT + t];
        aRow0 = emb + (size_t)tok * EE + shalf * 16;
    } else {
        aRow0 = x0 + (size_t)(m0 + srow) * K0 + shalf * 16;
    }
    const float* aRow1 = hprev + (size_t)(m0 + srow) * K1 + shalf * 16;
    // staging row -> weight column n = gate*512 + j0 + (row within 32)
    const int nGlob = ((srow >> 5) * HH) + j0 + (srow & 31);
    const unsigned short* bRow = WT + (size_t)nGlob * KTOT + shalf * 16;

    unsigned int* aStage = &As[srow * 16 + shalf * 8];   // this thread's slot
    unsigned int* bStage = &Bs[srow * 16 + shalf * 8];

    v8f acc[2][4];
    {
        const v8f vz = {0.f,0.f,0.f,0.f,0.f,0.f,0.f,0.f};
        #pragma unroll
        for (int i = 0; i < 2; ++i)
            #pragma unroll
            for (int g = 0; g < 4; ++g) acc[i][g] = vz;
    }

    // ---- prologue: stage chunk 0 into buffer 0 ----
    {
        const float* s = (0 < C0) ? aRow0 : aRow1;
        float4 fa[4];
        const float4* p = (const float4*)s;
        fa[0] = p[0]; fa[1] = p[1]; fa[2] = p[2]; fa[3] = p[3];
        pack_store_a(fa, aStage);
        const uint4* bp = (const uint4*)bRow;
        ((uint4*)bStage)[0] = bp[0];
        ((uint4*)bStage)[1] = bp[1];
    }
    __syncthreads();

    const unsigned int* aCur = As;        unsigned int* aNxtS = aStage + 2048;
    const unsigned int* bCur = Bs;        unsigned int* bNxtS = bStage + 2048;

    const int khalf = lane >> 4;
    const int rA    = wm * 32 + (lane & 15);

    #pragma unroll 2
    for (int kc = 0; kc < NCHUNK; ++kc) {
        // ---- (a) issue next chunk's global loads into registers ----
        float4 fa[4];
        uint4  wb0, wb1;
        const bool have = (kc + 1 < NCHUNK);
        if (have) {
            const int kn = kc + 1;
            const float* s = (kn < C0) ? aRow0 + kn * 32
                                       : aRow1 + (kn - C0) * 32;
            const float4* p = (const float4*)s;          // 64B aligned
            fa[0] = p[0]; fa[1] = p[1]; fa[2] = p[2]; fa[3] = p[3];
            const uint4* bp = (const uint4*)(bRow + kn * 32);
            wb0 = bp[0]; wb1 = bp[1];
            if (kc + 2 < NCHUNK)                         // global_prefetch_b8
                __builtin_prefetch(bRow + (kc + 2) * 32, 0, 1);
        }

        // ---- (b) compute current chunk from LDS[cur] ----
        FragU a[2], b[4];
        #pragma unroll
        for (int i = 0; i < 2; ++i) {
            const uint4* p = (const uint4*)&aCur[(rA + i * 16) * 16];
            uint4 lo = p[khalf];         // dwords khalf*4..+3   (K 0-7 / 8-15)
            uint4 hi = p[khalf + 2];     // dwords khalf*4+8..+11 (K 16-31)
            a[i].u[0]=lo.x; a[i].u[1]=lo.y; a[i].u[2]=lo.z; a[i].u[3]=lo.w;
            a[i].u[4]=hi.x; a[i].u[5]=hi.y; a[i].u[6]=hi.z; a[i].u[7]=hi.w;
        }
        #pragma unroll
        for (int g = 0; g < 4; ++g) {    // 4 gates of the same 16 j's
            const int rB = g * 32 + wn * 16 + (lane & 15);
            const uint4* p = (const uint4*)&bCur[rB * 16];
            uint4 lo = p[khalf];
            uint4 hi = p[khalf + 2];
            b[g].u[0]=lo.x; b[g].u[1]=lo.y; b[g].u[2]=lo.z; b[g].u[3]=lo.w;
            b[g].u[4]=hi.x; b[g].u[5]=hi.y; b[g].u[6]=hi.z; b[g].u[7]=hi.w;
        }
        #pragma unroll
        for (int i = 0; i < 2; ++i)
            #pragma unroll
            for (int g = 0; g < 4; ++g)
                acc[i][g] = __builtin_amdgcn_wmma_f32_16x16x32_bf16(
                    false, a[i].v, false, b[g].v,
                    (short)0, acc[i][g], false, false);

        // ---- (c) write next chunk into the other buffer ----
        if (have) {
            pack_store_a(fa, aNxtS);
            ((uint4*)bNxtS)[0] = wb0;
            ((uint4*)bNxtS)[1] = wb1;
        }
        __syncthreads();                 // one barrier per chunk

        // flip buffers
        const unsigned int* tc;
        unsigned int* ts;
        tc = aCur; aCur = (const unsigned int*)(aNxtS - (srow * 16 + shalf * 8));
        ts = aNxtS; aNxtS = (unsigned int*)tc + (srow * 16 + shalf * 8);
        tc = bCur; bCur = (const unsigned int*)(bNxtS - (srow * 16 + shalf * 8));
        ts = bNxtS; bNxtS = (unsigned int*)tc + (srow * 16 + shalf * 8);
        (void)ts;
    }

    // ---- fused LSTM cell epilogue ----
    // D layout: lane%16 = column (j within tile), vgpr g + 8*(lane>>4) = row.
    const int j  = j0 + wn * 16 + (lane & 15);
    const float bi = bias[j];
    const float bf = bias[j + HH];
    const float bg = bias[j + 2 * HH];
    const float bo = bias[j + 3 * HH];
    #pragma unroll
    for (int i = 0; i < 2; ++i)
        #pragma unroll
        for (int g = 0; g < 8; ++g) {
            const int m = m0 + wm * 32 + i * 16 + khalf * 8 + g;
            const size_t idx = (size_t)m * HH + j;
            const float zi = acc[i][0][g] + bi;
            const float zf = acc[i][1][g] + bf;
            const float zg = acc[i][2][g] + bg;
            const float zo = acc[i][3][g] + bo;
            const float cn = sigmoidf_(zf) * cState[idx]
                           + sigmoidf_(zi) * tanhf(zg);
            cState[idx] = cn;
            hOut[idx]   = sigmoidf_(zo) * tanhf(cn);
        }
}

// ---------------------------------------------------------------------------
// Build N-major bf16 [Wa;Wb]^T: outT[n*Ktot+k] = bf16(k<Ka ? Wa[k][n] : Wb[k-Ka][n])
// ---------------------------------------------------------------------------
__global__ __launch_bounds__(256)
void convert_wcat(const float* __restrict__ Wa, const float* __restrict__ Wb,
                  unsigned short* __restrict__ outT, int Ka, int Ktot)
{
    const int idx = blockIdx.x * blockDim.x + threadIdx.x;
    if (idx >= NB * Ktot) return;
    const int n = idx / Ktot;
    const int k = idx - n * Ktot;
    const float v = (k < Ka) ? Wa[(size_t)k * NB + n]
                             : Wb[(size_t)(k - Ka) * NB + n];
    unsigned int bits = __float_as_uint(v);
    bits = (bits + 0x7FFFu + ((bits >> 16) & 1u)) >> 16;
    outT[idx] = (unsigned short)bits;
}

__global__ void zero_f32(float* __restrict__ p, int n)
{
    const int i = blockIdx.x * blockDim.x + threadIdx.x;
    if (i < n) p[i] = 0.f;
}

__global__ void final_dense(const float* __restrict__ h2,
                            const float* __restrict__ Wfc,
                            const float* __restrict__ bfc,
                            float* __restrict__ out)
{
    const int b = blockIdx.x * blockDim.x + threadIdx.x;
    if (b >= BQ) return;
    float s = 0.f;
    for (int k = 0; k < HH; ++k) s += h2[(size_t)b * HH + k] * Wfc[k];
    out[b] = 1.f / (1.f + __expf(-(s + bfc[0])));
}

// ---------------------------------------------------------------------------
extern "C" void kernel_launch(void* const* d_in, const int* in_sizes, int n_in,
                              void* d_out, int out_size, void* d_ws, size_t ws_size,
                              hipStream_t stream)
{
    const int*   tokens = (const int*)  d_in[0];
    const float* emb    = (const float*)d_in[1];
    const float* W1     = (const float*)d_in[2];
    const float* U1     = (const float*)d_in[3];
    const float* b1     = (const float*)d_in[4];
    const float* W2     = (const float*)d_in[5];
    const float* U2     = (const float*)d_in[6];
    const float* b2     = (const float*)d_in[7];
    const float* Wfc    = (const float*)d_in[8];
    const float* bfc    = (const float*)d_in[9];
    float* out = (float*)d_out;

    // workspace carve-up (256B aligned): ~13 MB total
    char* ws = (char*)d_ws;
    size_t off = 0;
    auto carve = [&](size_t bytes) -> void* {
        void* p = ws + off;
        off = (off + bytes + 255) & ~(size_t)255;
        return p;
    };
    unsigned short* WT1 = (unsigned short*)carve((size_t)NB * 768  * 2); // [2048][768]
    unsigned short* WT2 = (unsigned short*)carve((size_t)NB * 1024 * 2); // [2048][1024]
    // zeroed together: c1, c2, h1[0], h2[0]
    float* zgrp = (float*)carve((size_t)4 * BQ * HH * 4);
    float* c1   = zgrp;
    float* c2   = zgrp + 1 * BQ * HH;
    float* h1a  = zgrp + 2 * BQ * HH;
    float* h2a  = zgrp + 3 * BQ * HH;
    float* h1b  = (float*)carve((size_t)BQ * HH * 4);
    float* h2b  = (float*)carve((size_t)BQ * HH * 4);
    float* h1buf[2] = { h1a, h1b };
    float* h2buf[2] = { h2a, h2b };

    // zero recurrent state (c1, c2, h1[0], h2[0])
    zero_f32<<<(4 * BQ * HH + 255) / 256, 256, 0, stream>>>(zgrp, 4 * BQ * HH);

    // one-time weight transpose+convert (L2-resident afterwards)
    convert_wcat<<<(NB * 768  + 255) / 256, 256, 0, stream>>>(W1, U1, WT1, 256, 768);
    convert_wcat<<<(NB * 1024 + 255) / 256, 256, 0, stream>>>(W2, U2, WT2, 512, 1024);

    const dim3 ggrid(HH / 32, BQ / 128);   // (16, 4) = 64 workgroups
    for (int t = 0; t < TT; ++t) {
        const int p = t & 1;
        // layer 1: gates([x_t, h1]) -> c1, h1'
        lstm_step<768, 256, true><<<ggrid, 256, 0, stream>>>(
            tokens, emb, nullptr, h1buf[p], WT1, b1, c1, h1buf[1 - p], t);
        // layer 2: gates([h1', h2]) -> c2, h2'
        lstm_step<1024, 512, false><<<ggrid, 256, 0, stream>>>(
            tokens, emb, h1buf[1 - p], h2buf[p], WT2, b2, c2, h2buf[1 - p], t);
    }

    // t = TT-1 (odd) wrote h2buf[0]
    final_dense<<<(BQ + 255) / 256, 256, 0, stream>>>(h2buf[0], Wfc, bfc, out);
}
